// MultiHeadAttention_68676527063388
// MI455X (gfx1250) — matrix operand order repaired and checked by reading, not compiled, not run on hardware
//
#include <hip/hip_runtime.h>

// MHA: B=8, T=1024, C=768, H=12, HS=64. fp32 in/out, f16 WMMA internally.

#define Bb 8
#define Tt 1024
#define Cc 768
#define Hh 12
#define HS 64

typedef _Float16 v16h __attribute__((ext_vector_type(16)));
typedef _Float16 v8h  __attribute__((ext_vector_type(8)));
typedef float    v8f  __attribute__((ext_vector_type(8)));

static __device__ inline v8f wmma32(v16h a, v16h b, v8f c) {
  // D = A(16x32 f16) * B(32x16 f16) + C(16x16 f32)
  return __builtin_amdgcn_wmma_f32_16x16x32_f16(false, a, false, b, (short)0, c, false, false);
}

// A-fragment (16x32 f16) load from row-major [M, K] memory; p = &row[m=lane&15][k_base]
// lanes 0-15 hold K 0..15; lanes 16-31 hold K 16..31 (contiguous, matches B)
static __device__ inline v16h load_afrag(const _Float16* p, int lane) {
  return *(const v16h*)(p + ((lane < 16) ? 0 : 16));
}

// B-fragment (32x16 f16) load; operand stored N-major [N, K]; p = &row[n=lane&15][k_base]
// lanes 0-15 hold K 0..15, lanes 16-31 hold K 16..31 (contiguous 32B per lane)
static __device__ inline v16h load_bfrag(const _Float16* p, int lane) {
  return *(const v16h*)(p + ((lane < 16) ? 0 : 16));
}

static __device__ inline float redmax16(float v) {
  v = fmaxf(v, __shfl_xor(v, 1));
  v = fmaxf(v, __shfl_xor(v, 2));
  v = fmaxf(v, __shfl_xor(v, 4));
  v = fmaxf(v, __shfl_xor(v, 8));
  return v;
}
static __device__ inline float redsum16(float v) {
  v += __shfl_xor(v, 1);
  v += __shfl_xor(v, 2);
  v += __shfl_xor(v, 4);
  v += __shfl_xor(v, 8);
  return v;
}

// ---------------- elementwise conversions ----------------
__global__ __launch_bounds__(256) void k_cvt_f16(const float* __restrict__ in,
                                                 _Float16* __restrict__ out, int n) {
  int i = blockIdx.x * 256 + threadIdx.x;
  if (i < n) out[i] = (_Float16)in[i];
}

// Wc[h][n][c], n in [0,192): n<64 -> Wq[h][c][n], n<128 -> Wk, else Wv (transposed to N-major)
__global__ __launch_bounds__(256) void k_build_wc(const float* __restrict__ Wq,
                                                  const float* __restrict__ Wk,
                                                  const float* __restrict__ Wv,
                                                  _Float16* __restrict__ Wc) {
  int i = blockIdx.x * 256 + threadIdx.x;     // i < 12*192*768
  int c = i % Cc;
  int n = (i / Cc) % 192;
  int h = i / (Cc * 192);
  float v;
  if (n < 64)       v = Wq[((size_t)h * Cc + c) * HS + n];
  else if (n < 128) v = Wk[((size_t)h * Cc + c) * HS + (n - 64)];
  else              v = Wv[((size_t)h * Cc + c) * HS + (n - 128)];
  Wc[i] = (_Float16)v;
}

// ---------------- QKV projection GEMM ----------------
// grid = B*H*8 blocks (128 query rows each), 256 threads (8 waves).
// Per wave: 32 rows x 96 cols (12 WMMA accumulators).
// Shared B slab (192 x 32 f16) is staged into LDS with async-to-LDS DMA,
// double-buffered across the K loop (ASYNCcnt + barrier).
// out: Qh/Kh row-major [bh][t][d] f16; Vt transposed [bh][d][t] f16
__global__ __launch_bounds__(256) void k_qkv(const _Float16* __restrict__ xh,
                                             const _Float16* __restrict__ Wc,
                                             _Float16* __restrict__ Qh,
                                             _Float16* __restrict__ Kh,
                                             _Float16* __restrict__ Vt) {
  __shared__ _Float16 bslab[2][192][32];   // 24 KB

  const int blk = blockIdx.x;
  const int mt = blk & 7;         // 128-row tile within T
  const int bh = blk >> 3;        // b*H + h
  const int b  = bh / Hh;
  const int h  = bh % Hh;
  const int tid = threadIdx.x, lane = tid & 31, w = tid >> 5;
  const int rquad = w & 3;        // 2 row-tiles per wave: rtile = 2*rquad + mi
  const int chalf = w >> 2;       // waves 0-3: cols 0..95, waves 4-7: 96..191
  const int lane16 = lane & 15;
  const int hi8 = (lane >> 4) * 8;

  const _Float16* wcbase = Wc + (size_t)h * 192 * Cc;

  // async stage of one 192x32 B slab (12 KB): 768 16B chunks over 256 threads
  auto issue_slab = [&](int kk, int bufi) {
#pragma unroll
    for (int q = 0; q < 3; ++q) {
      const int ch = q * 256 + tid;          // 0..767
      const int n = ch >> 2, sub = ch & 3;   // row, 16B chunk within 64B row
      const _Float16* g = wcbase + (size_t)n * Cc + kk + sub * 8;
      const unsigned l = (unsigned)(size_t)&bslab[bufi][n][sub * 8];
      asm volatile("global_load_async_to_lds_b128 %0, %1, off"
                   :: "v"(l), "v"(g) : "memory");
    }
  };

  const _Float16* arow0 = xh + ((size_t)b * Tt + mt * 128 + (2 * rquad + 0) * 16 + lane16) * Cc;
  const _Float16* arow1 = xh + ((size_t)b * Tt + mt * 128 + (2 * rquad + 1) * 16 + lane16) * Cc;

  v8f acc[2][6];
  const v8f z = {0.f,0.f,0.f,0.f,0.f,0.f,0.f,0.f};
#pragma unroll
  for (int mi = 0; mi < 2; ++mi)
#pragma unroll
    for (int j = 0; j < 6; ++j) acc[mi][j] = z;

  issue_slab(0, 0);
  asm volatile("s_wait_asynccnt 0" ::: "memory");
  __syncthreads();

  for (int ik = 0; ik < Cc / 32; ++ik) {
    const int kk = ik * 32;
    const int cur = ik & 1;
    if (ik + 1 < Cc / 32) issue_slab(kk + 32, cur ^ 1);

    v16h a0 = load_afrag(arow0 + kk, lane);
    v16h a1 = load_afrag(arow1 + kk, lane);
#pragma unroll
    for (int j = 0; j < 6; ++j) {
      v16h bf = load_bfrag(&bslab[cur][chalf * 96 + j * 16 + lane16][0], lane);
      acc[0][j] = wmma32(a0, bf, acc[0][j]);
      acc[1][j] = wmma32(a1, bf, acc[1][j]);
    }

    asm volatile("s_wait_asynccnt 0" ::: "memory");
    __syncthreads();
  }

#pragma unroll
  for (int mi = 0; mi < 2; ++mi) {
    const int tbase = mt * 128 + (2 * rquad + mi) * 16;
#pragma unroll
    for (int j = 0; j < 6; ++j) {
      const int nb = chalf * 96 + j * 16;
      if (nb < 64) {                 // Q, row-major [t][d]
        const int d = nb + lane16;
#pragma unroll
        for (int r = 0; r < 8; ++r) {
          const int t = tbase + r + hi8;
          Qh[((size_t)bh * Tt + t) * HS + d] = (_Float16)acc[mi][j][r];
        }
      } else if (nb < 128) {         // K, row-major [t][d]
        const int d = nb - 64 + lane16;
#pragma unroll
        for (int r = 0; r < 8; ++r) {
          const int t = tbase + r + hi8;
          Kh[((size_t)bh * Tt + t) * HS + d] = (_Float16)acc[mi][j][r];
        }
      } else {                       // V transposed: [d][t], 8 consecutive t -> one b128 store
        const int d = nb - 128 + lane16;
        const int tb = tbase + hi8;
        v8h hv;
#pragma unroll
        for (int r = 0; r < 8; ++r) hv[r] = (_Float16)acc[mi][j][r];
        *(v8h*)(Vt + ((size_t)bh * HS + d) * Tt + tb) = hv;
      }
    }
  }
}

// ---------------- flash attention (causal, online softmax) ----------------
// grid = B*H*16 blocks (64 q rows), 128 threads (4 waves, 16 q rows per wave)
__global__ __launch_bounds__(128) void k_attn(const _Float16* __restrict__ Qh,
                                              const _Float16* __restrict__ Kh,
                                              const _Float16* __restrict__ Vt,
                                              _Float16* __restrict__ AOh) {
  __shared__ _Float16 lds[4][16][32];   // per-wave P tile re-layout buffer
  const int blk = blockIdx.x;
  const int qt = blk & 15;
  const int bh = blk >> 4;
  const int b = bh / Hh, h = bh % Hh;
  const int tid = threadIdx.x, lane = tid & 31, w = tid >> 5;
  const int lane16 = lane & 15;
  const int hi8 = (lane >> 4) * 8;
  const int rowbase = qt * 64 + w * 16;

  const _Float16* qrow = Qh + ((size_t)bh * Tt + rowbase + lane16) * HS;
  const v16h aq0 = load_afrag(qrow, lane);        // d = 0..31
  const v16h aq1 = load_afrag(qrow + 32, lane);   // d = 32..63

  v8f o[4];
  const v8f z = {0.f,0.f,0.f,0.f,0.f,0.f,0.f,0.f};
#pragma unroll
  for (int j = 0; j < 4; ++j) o[j] = z;
  float mr[8], lr[8];
#pragma unroll
  for (int r = 0; r < 8; ++r) { mr[r] = -1e38f; lr[r] = 0.f; }

  const float SC = 0.125f * 1.44269504088896f;    // HS^-0.5 * log2(e)
  const int ktmax = (rowbase + 15) >> 5;          // last 32-key tile touching our rows

  for (int kt = 0; kt <= ktmax; ++kt) {
    const int sb = kt * 32;
    // ---- S = Q K^T for 16x32 key slab (two 16x16 score tiles) ----
    const _Float16* krow0 = Kh + ((size_t)bh * Tt + sb + lane16) * HS;
    const _Float16* krow1 = krow0 + (size_t)16 * HS;
    v8f s0 = z, s1 = z;
    s0 = wmma32(aq0, load_bfrag(krow0, lane), s0);
    s0 = wmma32(aq1, load_bfrag(krow0 + 32, lane), s0);
    s1 = wmma32(aq0, load_bfrag(krow1, lane), s1);
    s1 = wmma32(aq1, load_bfrag(krow1 + 32, lane), s1);

    const bool need_mask = (sb + 31) > rowbase;
    float p0[8], p1[8];
#pragma unroll
    for (int r = 0; r < 8; ++r) {
      float x0 = s0[r] * SC, x1 = s1[r] * SC;
      if (need_mask) {
        const int row = rowbase + r + hi8;
        if (sb + lane16 > row)      x0 = -1e38f;
        if (sb + 16 + lane16 > row) x1 = -1e38f;
      }
      float cand = redmax16(fmaxf(x0, x1));
      const float mnew = fmaxf(mr[r], cand);
      const float alpha = exp2f(mr[r] - mnew);
      p0[r] = exp2f(x0 - mnew);
      p1[r] = exp2f(x1 - mnew);
      const float rs = redsum16(p0[r] + p1[r]);
      lr[r] = lr[r] * alpha + rs;
      mr[r] = mnew;
      o[0][r] *= alpha; o[1][r] *= alpha; o[2][r] *= alpha; o[3][r] *= alpha;
    }

    // ---- P (C-layout f32) -> LDS row-major f16 -> A-fragment ----
#pragma unroll
    for (int r = 0; r < 8; ++r) {
      lds[w][r + hi8][lane16]      = (_Float16)p0[r];
      lds[w][r + hi8][16 + lane16] = (_Float16)p1[r];
    }
    asm volatile("s_wait_dscnt 0" ::: "memory");
    const v16h ap = load_afrag(&lds[w][lane16][0], lane);

    // ---- O += P * V  (V^T layout: [d][t], t contiguous) ----
#pragma unroll
    for (int j = 0; j < 4; ++j) {
      const _Float16* vrow = Vt + ((size_t)bh * HS + j * 16 + lane16) * Tt + sb;
      o[j] = wmma32(ap, load_bfrag(vrow, lane), o[j]);
    }
  }

  // ---- normalize and store f16 [b][t][h*64+d] ----
#pragma unroll
  for (int r = 0; r < 8; ++r) {
    const float inv = 1.0f / lr[r];
    const size_t trow = ((size_t)b * Tt + rowbase + r + hi8) * Cc + h * HS;
#pragma unroll
    for (int j = 0; j < 4; ++j) {
      AOh[trow + j * 16 + lane16] = (_Float16)(o[j][r] * inv);
    }
  }
}

// ---------------- output projection: out = AO @ Wproj^T + b ----------------
// grid = 64 row-blocks (128 rows) x 12 col-blocks (64 cols); 256 threads (8 waves)
__global__ __launch_bounds__(256) void k_proj(const _Float16* __restrict__ AOh,
                                              const _Float16* __restrict__ Wph,
                                              const float* __restrict__ bproj,
                                              float* __restrict__ out) {
  const int bm = blockIdx.x & 63;
  const int bn = blockIdx.x >> 6;
  const int tid = threadIdx.x, lane = tid & 31, w = tid >> 5;
  const int lane16 = lane & 15;
  const int hi8 = (lane >> 4) * 8;
  const int rb = bm * 128 + w * 16;

  const _Float16* arow = AOh + ((size_t)rb + lane16) * Cc;
  const _Float16* wbase = Wph + ((size_t)bn * 64 + lane16) * Cc;

  v8f acc[4];
  const v8f z = {0.f,0.f,0.f,0.f,0.f,0.f,0.f,0.f};
#pragma unroll
  for (int j = 0; j < 4; ++j) acc[j] = z;

  for (int kk = 0; kk < Cc; kk += 32) {
    v16h a = load_afrag(arow + kk, lane);
#pragma unroll
    for (int j = 0; j < 4; ++j) {
      v16h bf = load_bfrag(wbase + (size_t)j * 16 * Cc + kk, lane);
      acc[j] = wmma32(a, bf, acc[j]);
    }
  }

#pragma unroll
  for (int j = 0; j < 4; ++j) {
    const int c = bn * 64 + j * 16 + lane16;
    const float bj = bproj[c];
#pragma unroll
    for (int r = 0; r < 8; ++r) {
      out[((size_t)rb + r + hi8) * Cc + c] = acc[j][r] + bj;
    }
  }
}

extern "C" void kernel_launch(void* const* d_in, const int* in_sizes, int n_in,
                              void* d_out, int out_size, void* d_ws, size_t ws_size,
                              hipStream_t stream) {
  const float* x     = (const float*)d_in[0];
  const float* Wq    = (const float*)d_in[1];
  const float* Wk    = (const float*)d_in[2];
  const float* Wv    = (const float*)d_in[3];
  const float* Wproj = (const float*)d_in[4];
  const float* bproj = (const float*)d_in[5];
  float* out = (float*)d_out;

  char* ws = (char*)d_ws;
  const size_t N_X  = (size_t)Bb * Tt * Cc;        // 6291456
  const size_t N_WC = (size_t)Hh * 192 * Cc;       // 1769472
  const size_t N_WP = (size_t)Cc * Cc;             // 589824
  const size_t N_Q  = (size_t)Bb * Hh * Tt * HS;   // 6291456

  _Float16* xh  = (_Float16*)(ws);
  _Float16* Wc  = (_Float16*)(ws + 2 * N_X);
  _Float16* Wph = (_Float16*)(ws + 2 * (N_X + N_WC));
  _Float16* Qh  = (_Float16*)(ws + 2 * (N_X + N_WC + N_WP));
  _Float16* Kh  = (_Float16*)(ws + 2 * (N_X + N_WC + N_WP + N_Q));
  _Float16* Vt  = (_Float16*)(ws + 2 * (N_X + N_WC + N_WP + 2 * N_Q));
  _Float16* AOh = (_Float16*)(ws + 2 * (N_X + N_WC + N_WP + 3 * N_Q));

  k_cvt_f16<<<(int)(N_X / 256), 256, 0, stream>>>(x, xh, (int)N_X);
  k_build_wc<<<(int)(N_WC / 256), 256, 0, stream>>>(Wq, Wk, Wv, Wc);
  k_cvt_f16<<<(int)(N_WP / 256), 256, 0, stream>>>(Wproj, Wph, (int)N_WP);

  k_qkv<<<Bb * Hh * (Tt / 128), 256, 0, stream>>>(xh, Wc, Qh, Kh, Vt);
  k_attn<<<Bb * Hh * (Tt / 64), 128, 0, stream>>>(Qh, Kh, Vt, AOh);
  k_proj<<<(Bb * Tt / 128) * (Cc / 64), 256, 0, stream>>>(AOh, Wph, bproj, out);
}
